// SingleHeadAttention_57827439673739
// MI455X (gfx1250) — matrix operand order repaired and checked by reading, not compiled, not run on hardware
//
#include <hip/hip_runtime.h>

// SingleHeadAttention for MI455X (gfx1250, wave32, WMMA).
// fp32 -> f16 (hi,lo) split done ONCE into SoA planes laid out in WMMA operand
// order, so every fragment load is contiguous b128 with zero conversion VALU.
// 3x v_wmma_f32_16x16x32_f16 per tile gives fp32-class accuracy at f16 rate.

typedef _Float16 v16h __attribute__((ext_vector_type(16)));
typedef float    v8f  __attribute__((ext_vector_type(8)));
typedef _Float16 h2   __attribute__((ext_vector_type(2)));
typedef _Float16 h8   __attribute__((ext_vector_type(8)));

static constexpr int Bc = 4;
static constexpr int Sc = 2048;
static constexpr int Ec = 1024;
static constexpr int Ac = 128;

static __device__ __forceinline__ v8f wmma_f16(v16h a, v16h b, v8f c) {
  return __builtin_amdgcn_wmma_f32_16x16x32_f16(false, a, false, b, (short)0, c,
                                                false, false);
}

// (ah+al)*(bh+bl) ~= ah*bh + ah*bl + al*bh  (fp32-accurate f16 emulation)
static __device__ __forceinline__ v8f wmma3(v16h ah, v16h al, v16h bh, v16h bl, v8f c) {
  c = wmma_f16(ah, bh, c);
  c = wmma_f16(ah, bl, c);
  c = wmma_f16(al, bh, c);
  return c;
}

static __device__ __forceinline__ v16h mk16(h8 a, h8 b) {
  return __builtin_shufflevector(a, b, 0, 1, 2, 3, 4, 5, 6, 7, 8, 9, 10, 11, 12,
                                 13, 14, 15);
}

// A-operand (16x32): lane half hf holds K = hf*8..+7 and 16+hf*8..+7.
// p = row origin at the K-chunk base; both chunks are 16B aligned.
static __device__ __forceinline__ v16h loadA(const _Float16* p, int hf) {
  h8 c0 = *(const h8*)(p + hf * 8);
  h8 c1 = *(const h8*)(p + 16 + hf * 8);
  return mk16(c0, c1);
}

// B-operand (32x16): lane half hf holds K = hf*8..+7 and 16+hf*8..+7.
static __device__ __forceinline__ v16h loadB(const _Float16* p, int hf) {
  h8 c0 = *(const h8*)(p + hf * 8);
  h8 c1 = *(const h8*)(p + 16 + hf * 8);
  return mk16(c0, c1);
}

static __device__ __forceinline__ void split(float x, _Float16& h, _Float16& l) {
  h = (_Float16)x;
  l = (_Float16)(x - (float)h);
}

static __device__ __forceinline__ int a_kidx(int e, int hf) {
  return (e < 8) ? (hf * 8 + e) : (16 + hf * 8 + (e - 8));
}

// ---------------------------------------------------------------------------
// Phase 0: split fp32 -> separate hi/lo f16 planes (memory-bound, one pass).
// ---------------------------------------------------------------------------
__global__ __launch_bounds__(256) void split_kernel(const float* __restrict__ src,
                                                    _Float16* __restrict__ hi,
                                                    _Float16* __restrict__ lo,
                                                    int n8) {
  int i = blockIdx.x * 256 + threadIdx.x;
  if (i >= n8) return;
  const float4* s4 = (const float4*)src;
  float4 x0 = s4[(size_t)i * 2];
  float4 x1 = s4[(size_t)i * 2 + 1];
  float t[8] = {x0.x, x0.y, x0.z, x0.w, x1.x, x1.y, x1.z, x1.w};
  h8 vh, vl;
#pragma unroll
  for (int e = 0; e < 8; ++e) {
    _Float16 h, l;
    split(t[e], h, l);
    vh[e] = h;
    vl[e] = l;
  }
  *(h8*)(hi + (size_t)i * 8) = vh;
  *(h8*)(lo + (size_t)i * 8) = vl;
}

// ---------------------------------------------------------------------------
// Phase 1: q/k/v = X @ W^T. No LDS, no conversions in the K loop: all operand
// fragments are direct b128 SoA loads. grid=(M/128, 3), 8 waves x 16 rows.
//   q -> interleaved (hi,lo) h2, row-major [M][A]   (read once in phase 2)
//   k -> SoA planes, row-major [M][A]               (b128 B-operand loads)
//   v -> SoA planes, transposed [B][A][S]           (b128 B-operand loads,
//                                                    vectorized h8 stores)
// ---------------------------------------------------------------------------
__global__ __launch_bounds__(256) void proj_kernel(
    const _Float16* __restrict__ Xhi, const _Float16* __restrict__ Xlo,
    const _Float16* __restrict__ Whi, const _Float16* __restrict__ Wlo,
    h2* __restrict__ qh2, _Float16* __restrict__ khi, _Float16* __restrict__ klo,
    _Float16* __restrict__ vThi, _Float16* __restrict__ vTlo) {
  const int tid = threadIdx.x;
  const int m0 = blockIdx.x * 128;
  const int mat = blockIdx.y;
  const _Float16* Wh = Whi + (size_t)mat * Ac * Ec;
  const _Float16* Wl = Wlo + (size_t)mat * Ac * Ec;
  const int wave = tid >> 5, lane = tid & 31;
  const int r = lane & 15, hf = lane >> 4;
  const size_t xrow = (size_t)(m0 + wave * 16 + r) * Ec;

  v8f acc[8] = {};
  for (int k0 = 0; k0 < Ec; k0 += 32) {
    v16h ah = loadA(Xhi + xrow + k0, hf);
    v16h al = loadA(Xlo + xrow + k0, hf);
#pragma unroll
    for (int nt = 0; nt < 8; ++nt) {
      const size_t wrow = (size_t)(nt * 16 + r) * Ec + k0;
      v16h bh = loadB(Wh + wrow, hf);
      v16h bl = loadB(Wl + wrow, hf);
      acc[nt] = wmma3(ah, al, bh, bl, acc[nt]);
    }
  }

  if (mat == 2) {  // v: transposed [B][A][S], packed 16B stores per plane
    const int b = m0 / Sc;
    const int s0r = (m0 - b * Sc) + wave * 16 + hf * 8;
#pragma unroll
    for (int nt = 0; nt < 8; ++nt) {
      h8 vh, vl;
#pragma unroll
      for (int i = 0; i < 8; ++i) {
        _Float16 h, l;
        split(acc[nt][i], h, l);
        vh[i] = h;
        vl[i] = l;
      }
      const size_t addr = ((size_t)b * Ac + nt * 16 + r) * Sc + s0r;
      *(h8*)(vThi + addr) = vh;
      *(h8*)(vTlo + addr) = vl;
    }
  } else if (mat == 1) {  // k: SoA row-major planes
#pragma unroll
    for (int nt = 0; nt < 8; ++nt) {
#pragma unroll
      for (int i = 0; i < 8; ++i) {
        const size_t addr =
            (size_t)(m0 + wave * 16 + hf * 8 + i) * Ac + nt * 16 + r;
        _Float16 h, l;
        split(acc[nt][i], h, l);
        khi[addr] = h;
        klo[addr] = l;
      }
    }
  } else {  // q: fold softmax scale * log2(e); interleaved h2
    const float scale = 1.4426950408889634f / 11.313708498984761f;
#pragma unroll
    for (int nt = 0; nt < 8; ++nt) {
#pragma unroll
      for (int i = 0; i < 8; ++i) {
        const size_t addr =
            (size_t)(m0 + wave * 16 + hf * 8 + i) * Ac + nt * 16 + r;
        float x = acc[nt][i] * scale;
        _Float16 h, l;
        split(x, h, l);
        h2 p;
        p.x = h;
        p.y = l;
        qh2[addr] = p;
      }
    }
  }
}

// ---------------------------------------------------------------------------
// Phase 2: causal flash attention. grid=(S/128, B), 8 waves x 16 query rows.
// No barriers in the key loop; per-wave LDS bounce for P reshape, written
// pre-split into hi/lo half arrays so the read side is pure ds_load_b128.
// ---------------------------------------------------------------------------
__global__ __launch_bounds__(256) void attn_kernel(
    const h2* __restrict__ qh2, const _Float16* __restrict__ khi,
    const _Float16* __restrict__ klo, const _Float16* __restrict__ vThi,
    const _Float16* __restrict__ vTlo, float* __restrict__ out) {
  __shared__ __align__(16) _Float16 Phi[8][16 * 32];
  __shared__ __align__(16) _Float16 Plo[8][16 * 32];

  const int tid = threadIdx.x;
  const int wave = tid >> 5, lane = tid & 31;
  const int r = lane & 15, hf = lane >> 4;
  const int b = blockIdx.y;
  const int q0 = blockIdx.x * 128 + wave * 16;
  const size_t kbase = (size_t)b * Sc * Ac;  // halves, [S][A]
  const size_t vbase = (size_t)b * Ac * Sc;  // halves, [A][S]

  // Preload q fragments (one-time; unpack cost negligible).
  v16h qhF[4], qlF[4];
#pragma unroll
  for (int as = 0; as < 4; ++as) {
#pragma unroll
    for (int e = 0; e < 16; ++e) {
      h2 p = qh2[((size_t)b * Sc + q0 + r) * Ac + as * 32 + a_kidx(e, hf)];
      qhF[as][e] = p.x;
      qlF[as][e] = p.y;
    }
  }

  v8f O[8] = {};
  float m_i[8], l_i[8];
#pragma unroll
  for (int i = 0; i < 8; ++i) {
    m_i[i] = -1e30f;
    l_i[i] = 0.0f;
  }

  _Float16* PHw = &Phi[wave][0];
  _Float16* PLw = &Plo[wave][0];

  for (int kb = 0; kb <= q0 + 15; kb += 32) {
    if (kb + 32 < Sc) {  // prefetch next key block (global_prefetch_b8)
      __builtin_prefetch(khi + kbase + (size_t)(kb + 32 + r) * Ac, 0, 0);
      __builtin_prefetch(vThi + vbase + (size_t)(r)*Sc + kb + 32, 0, 0);
    }

    // scores: two 16x16 subtiles over 32 keys, K-dim = A in 4 steps of 32
    v8f s0 = {}, s1 = {};
#pragma unroll
    for (int as = 0; as < 4; ++as) {
      const size_t k0row = kbase + (size_t)(kb + r) * Ac + as * 32;
      const size_t k1row = kbase + (size_t)(kb + 16 + r) * Ac + as * 32;
      v16h bh = loadB(khi + k0row, hf);
      v16h bl = loadB(klo + k0row, hf);
      s0 = wmma3(qhF[as], qlF[as], bh, bl, s0);
      bh = loadB(khi + k1row, hf);
      bl = loadB(klo + k1row, hf);
      s1 = wmma3(qhF[as], qlF[as], bh, bl, s1);
    }

    // causal mask only on diagonal blocks (wave-uniform scalar branch)
    const bool diag = (kb + 31 > q0);
    float alpha[8];
#pragma unroll
    for (int i = 0; i < 8; ++i) {
      float a0 = s0[i], a1 = s1[i];
      if (diag) {
        int qa = q0 + hf * 8 + i;
        a0 = (kb + r <= qa) ? a0 : -1e30f;
        a1 = (kb + 16 + r <= qa) ? a1 : -1e30f;
      }
      float t = fmaxf(a0, a1);
      t = fmaxf(t, __shfl_xor(t, 1));
      t = fmaxf(t, __shfl_xor(t, 2));
      t = fmaxf(t, __shfl_xor(t, 4));
      t = fmaxf(t, __shfl_xor(t, 8));
      float mnew = fmaxf(m_i[i], t);
      float al = __builtin_amdgcn_exp2f(m_i[i] - mnew);
      float p0 = __builtin_amdgcn_exp2f(a0 - mnew);
      float p1 = __builtin_amdgcn_exp2f(a1 - mnew);
      float rs = p0 + p1;
      rs += __shfl_xor(rs, 1);
      rs += __shfl_xor(rs, 2);
      rs += __shfl_xor(rs, 4);
      rs += __shfl_xor(rs, 8);
      l_i[i] = l_i[i] * al + rs;
      m_i[i] = mnew;
      alpha[i] = al;
      _Float16 h, l;
      split(p0, h, l);
      PHw[(hf * 8 + i) * 32 + r] = h;
      PLw[(hf * 8 + i) * 32 + r] = l;
      split(p1, h, l);
      PHw[(hf * 8 + i) * 32 + 16 + r] = h;
      PLw[(hf * 8 + i) * 32 + 16 + r] = l;
    }

#pragma unroll
    for (int nt = 0; nt < 8; ++nt)
#pragma unroll
      for (int i = 0; i < 8; ++i) O[nt][i] *= alpha[i];

    // P fragments: pure ds_load_b128 (same-wave DS ops are in-order)
    v16h ph = loadA(PHw + r * 32, hf);
    v16h pl = loadA(PLw + r * 32, hf);

    // O += P @ V  (V from transposed SoA planes, contiguous b128 loads)
#pragma unroll
    for (int nt = 0; nt < 8; ++nt) {
      const size_t vrow = vbase + (size_t)(nt * 16 + r) * Sc + kb;
      v16h vh = loadB(vThi + vrow, hf);
      v16h vl = loadB(vTlo + vrow, hf);
      O[nt] = wmma3(ph, pl, vh, vl, O[nt]);
    }
  }

  // Normalize and store fp32 output.
#pragma unroll
  for (int nt = 0; nt < 8; ++nt) {
#pragma unroll
    for (int i = 0; i < 8; ++i) {
      float inv = 1.0f / l_i[i];
      out[((size_t)b * Sc + q0 + hf * 8 + i) * Ac + nt * 16 + r] =
          O[nt][i] * inv;
    }
  }
}

extern "C" void kernel_launch(void* const* d_in, const int* in_sizes, int n_in,
                              void* d_out, int out_size, void* d_ws,
                              size_t ws_size, hipStream_t stream) {
  (void)in_sizes;
  (void)n_in;
  (void)out_size;
  (void)ws_size;
  const float* X = (const float*)d_in[0];
  const float* Wq = (const float*)d_in[1];
  const float* Wk = (const float*)d_in[2];
  const float* Wv = (const float*)d_in[3];
  float* out = (float*)d_out;

  const size_t ME = (size_t)Bc * Sc * Ec;  // 8,388,608
  const size_t MA = (size_t)Bc * Sc * Ac;  // 1,048,576
  const size_t WE = (size_t)Ac * Ec;       // 131,072 per matrix

  _Float16* p = (_Float16*)d_ws;  // ~46 MB total, rewritten every call
  _Float16* Xhi = p;  p += ME;
  _Float16* Xlo = p;  p += ME;
  _Float16* Whi = p;  p += 3 * WE;
  _Float16* Wlo = p;  p += 3 * WE;
  h2* qh2 = (h2*)p;   p += 2 * MA;
  _Float16* khi = p;  p += MA;
  _Float16* klo = p;  p += MA;
  _Float16* vThi = p; p += MA;
  _Float16* vTlo = p; p += MA;

  split_kernel<<<dim3((unsigned)(ME / 8 / 256)), 256, 0, stream>>>(X, Xhi, Xlo,
                                                                   (int)(ME / 8));
  split_kernel<<<dim3((unsigned)(WE / 8 / 256)), 256, 0, stream>>>(
      Wq, Whi, Wlo, (int)(WE / 8));
  split_kernel<<<dim3((unsigned)(WE / 8 / 256)), 256, 0, stream>>>(
      Wk, Whi + WE, Wlo + WE, (int)(WE / 8));
  split_kernel<<<dim3((unsigned)(WE / 8 / 256)), 256, 0, stream>>>(
      Wv, Whi + 2 * WE, Wlo + 2 * WE, (int)(WE / 8));

  proj_kernel<<<dim3((Bc * Sc) / 128, 3), 256, 0, stream>>>(
      Xhi, Xlo, Whi, Wlo, qh2, khi, klo, vThi, vTlo);
  attn_kernel<<<dim3(Sc / 128, Bc), 256, 0, stream>>>(qh2, khi, klo, vThi, vTlo,
                                                      out);
}
